// GNNLayer_4011499454813
// MI455X (gfx1250) — matrix operand order repaired and checked by reading, not compiled, not run on hardware
//
#include <hip/hip_runtime.h>
#include <cstdint>

// ---------------------------------------------------------------------------
// GNN layer for MI455X (gfx1250), wave32, WMMA bf16 path + TDM staging.
// B=8, N=256, all feature dims = 64.
//
// Decomposition:
//   x1 = relu(x @ w_n2e + b_n2e)                      (tiny)
//   y  = x1 @ w_e2e[64:128,:]                         (tiny, precomputed, bf16)
//   W_new[b,i,j,:] = relu(W[b,i,j,:] @ w_e2e[0:64,:] + A[b,i,j]*y[b,j,:] + b_e2e)
//   e              = relu(W_new @ w_e2n + b_e2n)
//   x2[b,i,:]      = sum_j (A[b,i,j]/deg[b,i]) * e[b,i,j,:]
//   x_new = relu(concat(x,x2) @ w_n2n + b_n2n)        (tiny)
// ---------------------------------------------------------------------------

typedef __attribute__((ext_vector_type(16))) __bf16        v16bf;
typedef __attribute__((ext_vector_type(8)))  float         v8f;
typedef __attribute__((ext_vector_type(4)))  unsigned int  u32x4;
typedef __attribute__((ext_vector_type(8)))  int           i32x8;
typedef __attribute__((ext_vector_type(4)))  int           i32x4;
typedef __attribute__((ext_vector_type(4)))  float         f32x4;

#define BATCH 8
#define NN    256
#define FD    64
#define SROW  72   // padded LDS row stride in halfwords (144B -> bank spread)

__device__ __forceinline__ unsigned short f2bf(float f) {
    union { float f; unsigned int u; } c; c.f = f;
    unsigned int u = c.u;
    u += 0x7fffu + ((u >> 16) & 1u);          // round-to-nearest-even
    return (unsigned short)(u >> 16);
}
__device__ __forceinline__ float bf2f(unsigned short h) {
    union { unsigned int u; float f; } c; c.u = ((unsigned int)h) << 16;
    return c.f;
}

union BFrag { u32x4 u[2]; v16bf v; };

// ---------------------------------------------------------------------------
// Kernel A: x1 = relu(x @ w_n2e + b_n2e);  y = x1 @ w_e2e[64:,:]  (y -> bf16)
// ---------------------------------------------------------------------------
__global__ __launch_bounds__(64) void gnn_node_msg_kernel(
    const float* __restrict__ x, const float* __restrict__ w_n2e,
    const float* __restrict__ b_n2e, const float* __restrict__ w_e2e,
    unsigned short* __restrict__ y)
{
    __shared__ float sx[FD], sx1[FD];
    const int bn = blockIdx.x;
    const int f  = threadIdx.x;
    sx[f] = x[(size_t)bn * FD + f];
    __syncthreads();
    float a = b_n2e[f];
    #pragma unroll 8
    for (int k = 0; k < FD; ++k) a += sx[k] * w_n2e[k * FD + f];
    sx1[f] = fmaxf(a, 0.f);
    __syncthreads();
    float yv = 0.f;
    #pragma unroll 8
    for (int k = 0; k < FD; ++k) yv += sx1[k] * w_e2e[(FD + k) * FD + f];
    y[(size_t)bn * FD + f] = f2bf(yv);
}

// ---------------------------------------------------------------------------
// Kernel B: per (b,i) row: both edge GEMMs with bf16 WMMA, fused epilogues.
// 256 threads = 8 waves; wave w handles j-tiles {w, w+8}.
// y[b] tile staged into LDS by the Tensor Data Mover (wave 0 issues one DMA).
// ---------------------------------------------------------------------------
__global__ __launch_bounds__(256) void gnn_edge_kernel(
    const float* __restrict__ A,  const float* __restrict__ W,
    const float* __restrict__ w_e2e, const float* __restrict__ b_e2e,
    const float* __restrict__ w_e2n, const float* __restrict__ b_e2n,
    const unsigned short* __restrict__ y,
    float* __restrict__ Wnew_out, float* __restrict__ x2_out)
{
    __shared__ __align__(16) unsigned short sW  [NN][SROW];  // W[b,i,:,:] bf16
    __shared__ __align__(16) unsigned short sWn [NN][SROW];  // W_new bf16 (stage2 A)
    __shared__ __align__(16) unsigned short sWeT[FD][SROW];  // w_e2e[0:64]^T bf16
    __shared__ __align__(16) unsigned short sWnT[FD][SROW];  // w_e2n^T bf16
    __shared__ __align__(16) unsigned short sY  [NN][FD];    // y[b,:,:] bf16 (TDM dest)
    __shared__ float sA[NN];
    __shared__ float sRed[64];
    __shared__ float sInv;
    __shared__ float sBiasE[FD], sBiasN[FD];
    __shared__ float sX2p[8][FD];                            // per-wave partials

    const int bi   = blockIdx.x;            // b*256 + i
    const int tid  = threadIdx.x;
    const int lane = tid & 31;
    const int wave = tid >> 5;
    const int half = lane >> 4;             // 0: lanes 0-15, 1: lanes 16-31
    const int l15  = lane & 15;

    // ---- y[b] tile: 32KB contiguous bf16, moved by the TDM ----
    const unsigned short* ysrc = y + (size_t)(bi >> 8) * (NN * FD);
#if defined(__gfx1250__) && __has_builtin(__builtin_amdgcn_tensor_load_to_lds)
    if (wave == 0) {
        const uint64_t ga = (uint64_t)(uintptr_t)ysrc;
        const uint32_t la = (uint32_t)(uintptr_t)&sY[0][0];  // LDS byte offset
        const uint32_t ndw = NN * FD / 2;                    // 8192 dwords = 32KB
        u32x4 g0;
        g0.x = 1u;                                           // count=1 (valid user D#)
        g0.y = la;                                           // lds_addr
        g0.z = (uint32_t)ga;                                 // global_addr[31:0]
        g0.w = (uint32_t)((ga >> 32) & 0x1FFFFFFu) | (2u << 30); // addr[56:32] | type=2
        i32x8 g1;
        g1[0] = (int)(2u << 16);                             // data_size=4B, mask=0
        g1[1] = (int)((ndw & 0xFFFFu) << 16);                // tensor_dim0[15:0]
        g1[2] = (int)(((ndw >> 16) & 0xFFFFu) | (1u << 16)); // dim0[31:16] | tensor_dim1=1
        g1[3] = (int)((ndw & 0xFFFFu) << 16);                // tile_dim0 = ndw
        g1[4] = 1;                                           // tile_dim1=1, tile_dim2=0
        g1[5] = (int)ndw;                                    // tensor_dim0_stride
        g1[6] = 0;
        g1[7] = 0;
        i32x4 g2 = {0, 0, 0, 0};
        i32x4 g3 = {0, 0, 0, 0};
        i32x8 g4 = {0, 0, 0, 0, 0, 0, 0, 0};                 // unused trailing group
        // clang-23 / therock-10.0 lane: 6-arg form (g0,g1,g2,g3,g4,cpol)
        __builtin_amdgcn_tensor_load_to_lds(g0, g1, g2, g3, g4, 0);
#if __has_builtin(__builtin_amdgcn_s_wait_tensorcnt)
        __builtin_amdgcn_s_wait_tensorcnt(0);
#else
        asm volatile("s_wait_tensorcnt 0" ::: "memory");
#endif
    }
#else
    for (int idx = tid; idx < NN * FD / 2; idx += 256)       // fallback: manual copy
        ((uint32_t*)sY)[idx] = ((const uint32_t*)ysrc)[idx];
#endif

    // ---- cooperative staging: W row -> bf16 LDS ----
    const float* Wrow = W + (size_t)bi * (NN * FD);
    for (int idx = tid; idx < NN * FD / 4; idx += 256) {     // 4096 float4's
        const int row = idx >> 4;
        const int c4  = (idx & 15) << 2;
        f32x4 wv = __builtin_nontemporal_load((const f32x4*)(Wrow + row * FD + c4));
        #pragma unroll
        for (int q = 0; q < 4; ++q) sW[row][c4 + q] = f2bf(wv[q]);
    }
    for (int idx = tid; idx < FD * FD; idx += 256) {         // transpose weights
        const int k = idx >> 6, n = idx & 63;
        sWeT[n][k] = f2bf(w_e2e[k * FD + n]);                // top half of w_e2e
        sWnT[n][k] = f2bf(w_e2n[k * FD + n]);
    }
    sA[tid] = A[(size_t)bi * NN + tid];
    if (tid < FD) { sBiasE[tid] = b_e2e[tid]; sBiasN[tid] = b_e2n[tid]; }
    for (int idx = tid; idx < 8 * FD; idx += 256) ((float*)sX2p)[idx] = 0.f;
    __syncthreads();
    if (tid < 64) {                                          // deg = sum_j A[i,j]
        float s = 0.f;
        #pragma unroll
        for (int q = 0; q < 4; ++q) s += sA[tid * 4 + q];
        sRed[tid] = s;
    }
    __syncthreads();
    if (tid == 0) {
        float d = 0.f;
        for (int q = 0; q < 64; ++q) d += sRed[q];
        sInv = 1.f / fmaxf(d, 1e-12f);
    }
    __syncthreads();

    const int aoff = half ? 8 : 0;    // A-frag: lanes>=16 hold K 8..15 / 24..31
    const int boff = half ? 8 : 0;    // B-frag: lanes>=16 hold K 8..15 / 24..31

    // ---- stage 1: W_new = relu(W @ w_e2e_top + A*y + b_e2e) ----
    for (int t = 0; t < 2; ++t) {
        const int jbase = (wave + t * 8) * 16;
        BFrag a0, a1;                                        // K 0..31 / 32..63
        {
            const int row = jbase + l15;
            a0.u[0] = *(const u32x4*)&sW[row][aoff];
            a0.u[1] = *(const u32x4*)&sW[row][aoff + 16];
            a1.u[0] = *(const u32x4*)&sW[row][32 + aoff];
            a1.u[1] = *(const u32x4*)&sW[row][32 + aoff + 16];
        }
        #pragma unroll
        for (int nt = 0; nt < 4; ++nt) {
            const int rowN = nt * 16 + l15;
            BFrag b0, b1;
            b0.u[0] = *(const u32x4*)&sWeT[rowN][boff];
            b0.u[1] = *(const u32x4*)&sWeT[rowN][boff + 16];
            b1.u[0] = *(const u32x4*)&sWeT[rowN][32 + boff];
            b1.u[1] = *(const u32x4*)&sWeT[rowN][32 + boff + 16];
            v8f acc = {};
            acc = __builtin_amdgcn_wmma_f32_16x16x32_bf16(false, a0.v, false, b0.v,
                                                          (short)0, acc, false, false);
            acc = __builtin_amdgcn_wmma_f32_16x16x32_bf16(false, a1.v, false, b1.v,
                                                          (short)0, acc, false, false);
            const int Ncol = nt * 16 + l15;
            const float bias = sBiasE[Ncol];
            #pragma unroll
            for (int r = 0; r < 8; ++r) {
                const int j = jbase + r + half * 8;
                float v = acc[r] + bias + sA[j] * bf2f(sY[j][Ncol]);
                v = fmaxf(v, 0.f);
                __builtin_nontemporal_store(v, &Wnew_out[((size_t)bi * NN + j) * FD + Ncol]);
                sWn[j][Ncol] = f2bf(v);
            }
        }
    }
    __syncthreads();

    // ---- stage 2: e = relu(W_new @ w_e2n + b_e2n); x2 += An * e ----
    for (int t = 0; t < 2; ++t) {
        const int jbase = (wave + t * 8) * 16;
        BFrag a0, a1;
        {
            const int row = jbase + l15;
            a0.u[0] = *(const u32x4*)&sWn[row][aoff];
            a0.u[1] = *(const u32x4*)&sWn[row][aoff + 16];
            a1.u[0] = *(const u32x4*)&sWn[row][32 + aoff];
            a1.u[1] = *(const u32x4*)&sWn[row][32 + aoff + 16];
        }
        #pragma unroll
        for (int nt = 0; nt < 4; ++nt) {
            const int rowN = nt * 16 + l15;
            BFrag b0, b1;
            b0.u[0] = *(const u32x4*)&sWnT[rowN][boff];
            b0.u[1] = *(const u32x4*)&sWnT[rowN][boff + 16];
            b1.u[0] = *(const u32x4*)&sWnT[rowN][32 + boff];
            b1.u[1] = *(const u32x4*)&sWnT[rowN][32 + boff + 16];
            v8f acc = {};
            acc = __builtin_amdgcn_wmma_f32_16x16x32_bf16(false, a0.v, false, b0.v,
                                                          (short)0, acc, false, false);
            acc = __builtin_amdgcn_wmma_f32_16x16x32_bf16(false, a1.v, false, b1.v,
                                                          (short)0, acc, false, false);
            const int Ncol = nt * 16 + l15;
            const float bias = sBiasN[Ncol];
            float p = 0.f;
            #pragma unroll
            for (int r = 0; r < 8; ++r) {
                const int j = jbase + r + half * 8;
                const float e = fmaxf(acc[r] + bias, 0.f);
                p += sA[j] * e;                              // 1/deg applied below
            }
            p *= sInv;
            p += __shfl_xor(p, 16, 32);                      // fold lane halves
            if (lane < 16) sX2p[wave][Ncol] += p;            // 1 writer / slot
        }
    }
    __syncthreads();

    if (tid < FD) {
        float s = 0.f;
        #pragma unroll
        for (int w8 = 0; w8 < 8; ++w8) s += sX2p[w8][tid];
        x2_out[(size_t)bi * FD + tid] = s;
    }
}

// ---------------------------------------------------------------------------
// Kernel C: x_new = relu(concat(x, x2) @ w_n2n + b_n2n)
// ---------------------------------------------------------------------------
__global__ __launch_bounds__(64) void gnn_node_update_kernel(
    const float* __restrict__ x, const float* __restrict__ x2,
    const float* __restrict__ w_n2n, const float* __restrict__ b_n2n,
    float* __restrict__ xnew)
{
    __shared__ float sx[FD], sx2[FD];
    const int bn = blockIdx.x;
    const int f  = threadIdx.x;
    sx[f]  = x [(size_t)bn * FD + f];
    sx2[f] = x2[(size_t)bn * FD + f];
    __syncthreads();
    float a = b_n2n[f];
    #pragma unroll 8
    for (int k = 0; k < FD; ++k) a += sx[k]  * w_n2n[k * FD + f];
    #pragma unroll 8
    for (int k = 0; k < FD; ++k) a += sx2[k] * w_n2n[(FD + k) * FD + f];
    xnew[(size_t)bn * FD + f] = fmaxf(a, 0.f);
}

// ---------------------------------------------------------------------------
extern "C" void kernel_launch(void* const* d_in, const int* in_sizes, int n_in,
                              void* d_out, int out_size, void* d_ws, size_t ws_size,
                              hipStream_t stream)
{
    const float* A     = (const float*)d_in[0];
    const float* W     = (const float*)d_in[1];
    const float* x     = (const float*)d_in[2];
    const float* w_n2e = (const float*)d_in[3];
    const float* b_n2e = (const float*)d_in[4];
    const float* w_e2e = (const float*)d_in[5];
    const float* b_e2e = (const float*)d_in[6];
    const float* w_e2n = (const float*)d_in[7];
    const float* b_e2n = (const float*)d_in[8];
    const float* w_n2n = (const float*)d_in[9];
    const float* b_n2n = (const float*)d_in[10];

    float* Wnew = (float*)d_out;                                   // [8,256,256,64]
    float* xnew = (float*)d_out + (size_t)BATCH * NN * NN * FD;    // [8,256,64]
    unsigned short* y = (unsigned short*)d_ws;                     // [8,256,64] bf16
    float* x2 = (float*)((char*)d_ws +
                         (size_t)BATCH * NN * FD * sizeof(unsigned short));

    gnn_node_msg_kernel<<<BATCH * NN, 64, 0, stream>>>(x, w_n2e, b_n2e, w_e2e, y);
    gnn_edge_kernel<<<BATCH * NN, 256, 0, stream>>>(A, W, w_e2e, b_e2e,
                                                    w_e2n, b_e2n, y, Wnew, x2);
    gnn_node_update_kernel<<<BATCH * NN, 64, 0, stream>>>(x, x2, w_n2n, b_n2n, xnew);
}
